// Attention_24988119728681
// MI455X (gfx1250) — matrix operand order repaired and checked by reading, compiled, not run on hardware
//
#include <hip/hip_runtime.h>
#include <hip/hip_bf16.h>
#include <stdint.h>

typedef __attribute__((ext_vector_type(16))) __bf16 v16bf;
typedef __attribute__((ext_vector_type(8)))  float  v8f;
typedef __attribute__((ext_vector_type(4)))  unsigned int u32x4;
typedef __attribute__((ext_vector_type(8)))  int i32x8;
typedef __attribute__((ext_vector_type(4)))  int i32x4;

union Frag {
  uint4 q[2];
  v16bf v;
};
union U4S8 { uint4 q; unsigned short s[8]; };

constexpr int CB = 2, CS = 2048, CHID = 2048, CNH = 32, CNKV = 8, CHD = 64;

__device__ __forceinline__ unsigned short f2bf(float f) {
  union { float f; unsigned u; } x; x.f = f;
  unsigned r = x.u + 0x7FFFu + ((x.u >> 16) & 1u);  // round-to-nearest-even
  return (unsigned short)(r >> 16);
}

__device__ __forceinline__ v8f wmma_bf16(v16bf a, v16bf b, v8f c) {
  return __builtin_amdgcn_wmma_f32_16x16x32_bf16(false, a, false, b, (short)0, c, false, false);
}

__device__ __forceinline__ float rmax16(float x) {
#pragma unroll
  for (int m = 1; m < 16; m <<= 1) x = fmaxf(x, __shfl_xor(x, m, 16));
  return x;
}
__device__ __forceinline__ float rsum16(float x) {
#pragma unroll
  for (int m = 1; m < 16; m <<= 1) x += __shfl_xor(x, m, 16);
  return x;
}

// Raw LDS byte offset: generic LDS addresses are {aperture_hi, offset32}.
__device__ __forceinline__ unsigned lds_raw(const void* p) {
  return (unsigned)(uintptr_t)p;
}

// ---------------------------------------------------------------------------
// fp32 -> bf16 flat convert
// ---------------------------------------------------------------------------
__global__ __launch_bounds__(256) void cvt_bf16(const float* __restrict__ src,
                                                unsigned short* __restrict__ dst, int total) {
  int idx = blockIdx.x * blockDim.x + threadIdx.x;
  if (idx < total) dst[idx] = f2bf(src[idx]);
}

// ---------------------------------------------------------------------------
// W[K][N] f32  ->  Wt[N][K] bf16  (LDS-tiled 32x32 transpose)
// ---------------------------------------------------------------------------
__global__ __launch_bounds__(256) void transpose_cvt(const float* __restrict__ W,
                                                     unsigned short* __restrict__ Wt,
                                                     int K, int N) {
  __shared__ unsigned short t[32][33];
  int k0 = blockIdx.y * 32, n0 = blockIdx.x * 32;
  int tx = threadIdx.x, ty = threadIdx.y;  // blockDim = (32, 8)
#pragma unroll
  for (int r = ty; r < 32; r += 8)
    t[r][tx] = f2bf(W[(size_t)(k0 + r) * N + n0 + tx]);
  __syncthreads();
#pragma unroll
  for (int r = ty; r < 32; r += 8)
    Wt[(size_t)(n0 + r) * K + k0 + tx] = t[tx][r];
}

// ---------------------------------------------------------------------------
// GEMM: C[M,N] = A[M,K] (bf16 row-major) * Bt[N,K]^T (bf16, pre-transposed).
// Block tile 128x64, K-step 32. A tile staged by TDM (tensor_load_to_lds),
// B tile staged by per-lane global_load_async_to_lds_b128.
// 8 waves: 4(M) x 2(N), each wave 32x32 = 2x2 WMMA tiles.
// ---------------------------------------------------------------------------
template <bool OUT_BF16>
__global__ __launch_bounds__(256) void gemm_bf16w(const unsigned short* __restrict__ A,
                                                  const unsigned short* __restrict__ Bt,
                                                  void* __restrict__ Cout,
                                                  int M, int N, int K) {
  __shared__ __align__(16) unsigned short As[128 * 32];  // [m][k]
  __shared__ __align__(16) unsigned short Bs[64 * 32];   // [n][k]

  const int tid  = threadIdx.x;
  const int wave = tid >> 5, lane = tid & 31;
  const int hi = lane >> 4, ln = lane & 15;
  const int waveM = wave >> 1, waveN = wave & 1;
  const int bm = blockIdx.y * 128, bn = blockIdx.x * 64;

  const v8f vzero = {0.f, 0.f, 0.f, 0.f, 0.f, 0.f, 0.f, 0.f};
  v8f acc[2][2] = {{vzero, vzero}, {vzero, vzero}};

  // B staging: 256 chunks of 16B (64 rows x 4 chunks), one per lane.
  const int c = wave * 32 + lane;
  const int bnrow = c >> 2, bpart = c & 3;
  const unsigned ldsB = lds_raw(&Bs[c * 8]);
  const unsigned ldsA = lds_raw(&As[0]);
  const i32x4 z4 = {0, 0, 0, 0};
  const i32x8 z8 = {0, 0, 0, 0, 0, 0, 0, 0};

  for (int k0 = 0; k0 < K; k0 += 32) {
    // ---- B tile: async copy, tracked by ASYNCcnt
    const unsigned short* bsrc = Bt + (size_t)(bn + bnrow) * K + k0 + bpart * 8;
    asm volatile("global_load_async_to_lds_b128 %0, %1, off"
                 :: "v"(ldsB), "v"(bsrc) : "memory");

    // ---- A tile: Tensor Data Mover 2D descriptor load (wave 0 issues)
    if (wave == 0) {
      unsigned long long ga = (unsigned long long)(uintptr_t)(A + (size_t)bm * K + k0);
      u32x4 g0;
      g0[0] = 1u;                                         // count=1, user descriptor
      g0[1] = ldsA;                                       // lds_addr
      g0[2] = (unsigned)ga;                               // global_addr[31:0]
      g0[3] = (unsigned)((ga >> 32) & 0x01FFFFFFu) | 0x80000000u;  // addr[56:32] | type=2
      i32x8 g1;
      g1[0] = 0x00010000;                                 // data_size = 1 (2 bytes)
      g1[1] = (int)((K & 0xFFFF) << 16);                  // tensor_dim0[15:0]
      g1[2] = (int)(((unsigned)K >> 16) | ((M & 0xFFFF) << 16));   // dim0 hi | dim1 lo
      g1[3] = (int)(((unsigned)M >> 16) | (32u << 16));   // dim1 hi | tile_dim0=32
      g1[4] = 128;                                        // tile_dim1 = 128
      g1[5] = (int)K;                                     // tensor_dim0_stride[31:0]
      g1[6] = 0;
      g1[7] = 0;
      __builtin_amdgcn_tensor_load_to_lds(g0, g1, z4, z4, z8, 0);
      __builtin_amdgcn_s_wait_tensorcnt(0);
    }
    asm volatile("s_wait_asynccnt 0" ::: "memory");
    __syncthreads();

    Frag a[2], b[2];
#pragma unroll
    for (int t = 0; t < 2; ++t) {  // A fragments: lane row M=ln
      int mr = (waveM * 32 + t * 16 + ln) * 32;
      a[t].q[0] = *(const uint4*)&As[mr + 8 * hi];       // elems 0..7 : K=8*hi..
      a[t].q[1] = *(const uint4*)&As[mr + 16 + 8 * hi];  // elems 8..15: K=16+8*hi..
    }
#pragma unroll
    for (int t = 0; t < 2; ++t) {  // B fragments: lane col N=ln, K=8*hi.. / 16+8*hi..
      int nr = (waveN * 32 + t * 16 + ln) * 32;
      b[t].q[0] = *(const uint4*)&Bs[nr + 8 * hi];
      b[t].q[1] = *(const uint4*)&Bs[nr + 16 + 8 * hi];
    }
#pragma unroll
    for (int i = 0; i < 2; ++i)
#pragma unroll
      for (int j = 0; j < 2; ++j) acc[i][j] = wmma_bf16(a[i].v, b[j].v, acc[i][j]);
    __syncthreads();
  }

#pragma unroll
  for (int i = 0; i < 2; ++i)
#pragma unroll
    for (int j = 0; j < 2; ++j)
#pragma unroll
      for (int r = 0; r < 8; ++r) {
        int row = bm + waveM * 32 + i * 16 + r + 8 * hi;
        int col = bn + waveN * 32 + j * 16 + ln;
        float val = acc[i][j][r];
        if (OUT_BF16)
          ((unsigned short*)Cout)[(size_t)row * N + col] = f2bf(val);
        else
          ((float*)Cout)[(size_t)row * N + col] = val;
      }
}

// ---------------------------------------------------------------------------
// RoPE + fp32 -> bf16 conversion. src: [B*S, nh*64] f32, dst same shape bf16.
// ---------------------------------------------------------------------------
__global__ __launch_bounds__(256) void rope_cvt(const float* __restrict__ src,
                                                const float* __restrict__ cosT,
                                                const float* __restrict__ sinT,
                                                unsigned short* __restrict__ dst,
                                                int nheads, int total) {
  int idx = blockIdx.x * blockDim.x + threadIdx.x;
  if (idx >= total) return;
  int cols = nheads * CHD;
  int row = idx / cols;
  int c = idx - row * cols;
  int d = c & (CHD - 1);
  int s = row & (CS - 1);
  float v = src[idx];
  float other = (d < 32) ? -src[idx + 32] : src[idx - 32];
  v = v * cosT[s * CHD + d] + other * sinT[s * CHD + d];
  dst[idx] = f2bf(v);
}

// ---------------------------------------------------------------------------
// Flash attention (causal, GQA 4:1). Each wave: one 16-row Q tile of one head.
// Keys processed 32 at a time: 4 score WMMAs + 4 P*V WMMAs per step.
// Output written as bf16 (feeds the Wo GEMM directly).
// ---------------------------------------------------------------------------
__global__ __launch_bounds__(256) void attn_flash(const unsigned short* __restrict__ Qb,
                                                  const unsigned short* __restrict__ Kb,
                                                  const unsigned short* __restrict__ Vb,
                                                  unsigned short* __restrict__ Outb) {
  __shared__ __align__(16) unsigned short Pl[8][16 * 32];  // per-wave P tile (bf16)
  __shared__ __align__(16) unsigned short Vt[8][64 * 32];  // per-wave V^T tile (bf16)

  const int tid = threadIdx.x;
  const int wave = tid >> 5, lane = tid & 31;
  const int hi = lane >> 4, ln = lane & 15;
  const int b = blockIdx.z, h = blockIdx.y, g = h >> 2;
  const int q0 = blockIdx.x * 128 + wave * 16;

  Frag qa[2];
  {
    const unsigned short* qrow = Qb + ((size_t)(b * CS + q0 + ln) * CNH + h) * CHD;
#pragma unroll
    for (int c = 0; c < 2; ++c) {
      qa[c].q[0] = *(const uint4*)(qrow + 32 * c + 8 * hi);
      qa[c].q[1] = *(const uint4*)(qrow + 32 * c + 16 + 8 * hi);
    }
  }

  const v8f vzero = {0.f, 0.f, 0.f, 0.f, 0.f, 0.f, 0.f, 0.f};
  v8f o[4] = {vzero, vzero, vzero, vzero};
  float mrow[8], lrow[8];
#pragma unroll
  for (int r = 0; r < 8; ++r) { mrow[r] = -1e30f; lrow[r] = 0.f; }

  for (int kb = 0; kb < q0 + 16; kb += 32) {
    Frag k0f[2], k1f[2];
    const unsigned short* kr0 = Kb + ((size_t)(b * CS + kb + ln) * CNKV + g) * CHD;
    const unsigned short* kr1 = kr0 + (size_t)16 * CNKV * CHD;
#pragma unroll
    for (int c = 0; c < 2; ++c) {
      k0f[c].q[0] = *(const uint4*)(kr0 + 32 * c + 8 * hi);
      k0f[c].q[1] = *(const uint4*)(kr0 + 32 * c + 16 + 8 * hi);
      k1f[c].q[0] = *(const uint4*)(kr1 + 32 * c + 8 * hi);
      k1f[c].q[1] = *(const uint4*)(kr1 + 32 * c + 16 + 8 * hi);
    }
    v8f s0 = vzero, s1 = vzero;
    s0 = wmma_bf16(qa[0].v, k0f[0].v, s0);
    s0 = wmma_bf16(qa[1].v, k0f[1].v, s0);
    s1 = wmma_bf16(qa[0].v, k1f[0].v, s1);
    s1 = wmma_bf16(qa[1].v, k1f[1].v, s1);

    // Stage V tile transposed into LDS: Vt[d][key], key = lane.
    {
      const unsigned short* vr = Vb + ((size_t)(b * CS + kb + lane) * CNKV + g) * CHD;
#pragma unroll
      for (int d0 = 0; d0 < 64; d0 += 8) {
        U4S8 u; u.q = *(const uint4*)(vr + d0);
#pragma unroll
        for (int e = 0; e < 8; ++e) Vt[wave][(d0 + e) * 32 + lane] = u.s[e];
      }
    }

    const float scale = 0.125f;  // 1/sqrt(64)
    float p0v[8], p1v[8];
#pragma unroll
    for (int r = 0; r < 8; ++r) {
      int sq = q0 + r + 8 * hi;
      int sk0 = kb + ln, sk1 = sk0 + 16;
      float a0 = (sk0 <= sq) ? s0[r] * scale : -1e30f;
      float a1 = (sk1 <= sq) ? s1[r] * scale : -1e30f;
      float mt = rmax16(fmaxf(a0, a1));
      float mn = fmaxf(mrow[r], mt);
      float al = __expf(mrow[r] - mn);
      mrow[r] = mn;
      float p0 = __expf(a0 - mn), p1 = __expf(a1 - mn);
      lrow[r] = lrow[r] * al + rsum16(p0 + p1);
      o[0][r] *= al; o[1][r] *= al; o[2][r] *= al; o[3][r] *= al;
      p0v[r] = p0; p1v[r] = p1;
    }

    // P (C/D layout) -> LDS -> A-operand layout.
#pragma unroll
    for (int r = 0; r < 8; ++r) {
      Pl[wave][(r + 8 * hi) * 32 + ln]      = f2bf(p0v[r]);
      Pl[wave][(r + 8 * hi) * 32 + 16 + ln] = f2bf(p1v[r]);
    }
    __builtin_amdgcn_wave_barrier();
    asm volatile("s_wait_dscnt 0" ::: "memory");
    __builtin_amdgcn_wave_barrier();

    Frag pa;
    pa.q[0] = *(const uint4*)&Pl[wave][ln * 32 + 8 * hi];
    pa.q[1] = *(const uint4*)&Pl[wave][ln * 32 + 16 + 8 * hi];
#pragma unroll
    for (int j = 0; j < 4; ++j) {
      Frag vb;
      vb.q[0] = *(const uint4*)&Vt[wave][(j * 16 + ln) * 32 + 8 * hi];
      vb.q[1] = *(const uint4*)&Vt[wave][(j * 16 + ln) * 32 + 16 + 8 * hi];
      o[j] = wmma_bf16(pa.v, vb.v, o[j]);
    }
  }

#pragma unroll
  for (int r = 0; r < 8; ++r) {
    float inv = 1.0f / lrow[r];
    int sq = q0 + r + 8 * hi;
    unsigned short* orow = Outb + ((size_t)(b * CS + sq) * CNH + h) * CHD;
#pragma unroll
    for (int j = 0; j < 4; ++j) orow[j * 16 + ln] = f2bf(o[j][r] * inv);
  }
}

// ---------------------------------------------------------------------------
extern "C" void kernel_launch(void* const* d_in, const int* in_sizes, int n_in,
                              void* d_out, int out_size, void* d_ws, size_t ws_size,
                              hipStream_t stream) {
  (void)in_sizes; (void)n_in; (void)out_size; (void)ws_size;
  const float* x    = (const float*)d_in[0];
  const float* cosT = (const float*)d_in[1];
  const float* sinT = (const float*)d_in[2];
  const float* Wq   = (const float*)d_in[3];
  const float* Wk   = (const float*)d_in[4];
  const float* Wv   = (const float*)d_in[5];
  const float* Wo   = (const float*)d_in[6];
  float* out = (float*)d_out;
  char* ws = (char*)d_ws;

  const int M = CB * CS;  // 4096
  // Workspace layout (100 MB):
  unsigned short* xb  = (unsigned short*)(ws);              // 16 MB  x in bf16
  unsigned short* Wqt = (unsigned short*)(ws + 16777216);   //  8 MB  Wq^T bf16
  unsigned short* Wkt = (unsigned short*)(ws + 25165824);   //  2 MB
  unsigned short* Wvt = (unsigned short*)(ws + 27262976);   //  2 MB
  unsigned short* Wot = (unsigned short*)(ws + 29360128);   //  8 MB
  float*          qf  = (float*)(ws + 37748736);            // 32 MB  q f32 (reused: attn out bf16)
  float*          kf  = (float*)(ws + 71303168);            //  8 MB  k f32
  unsigned short* qb  = (unsigned short*)(ws + 79691776);   // 16 MB
  unsigned short* kb  = (unsigned short*)(ws + 96468992);   //  4 MB
  unsigned short* vb  = (unsigned short*)(ws + 100663296);  //  4 MB
  unsigned short* ab  = (unsigned short*)qf;                // attn out bf16 (16 MB within qf)

  dim3 blk(256);
  // Pre-convert activations / weights to bf16 (weights transposed to [N][K]).
  cvt_bf16<<<(M * CHID) / 256, blk, 0, stream>>>(x, xb, M * CHID);
  dim3 tblk(32, 8);
  transpose_cvt<<<dim3(CHID / 32, CHID / 32), tblk, 0, stream>>>(Wq, Wqt, CHID, CHID);
  transpose_cvt<<<dim3((CNKV * CHD) / 32, CHID / 32), tblk, 0, stream>>>(Wk, Wkt, CHID, CNKV * CHD);
  transpose_cvt<<<dim3((CNKV * CHD) / 32, CHID / 32), tblk, 0, stream>>>(Wv, Wvt, CHID, CNKV * CHD);
  transpose_cvt<<<dim3(CHID / 32, CHID / 32), tblk, 0, stream>>>(Wo, Wot, CHID, CHID);

  // QKV projections (bf16 WMMA, TDM + async LDS staging)
  gemm_bf16w<false><<<dim3(CHID / 64, M / 128), blk, 0, stream>>>(xb, Wqt, qf, M, CHID, CHID);
  gemm_bf16w<false><<<dim3((CNKV * CHD) / 64, M / 128), blk, 0, stream>>>(xb, Wkt, kf, M, CNKV * CHD, CHID);
  gemm_bf16w<true><<<dim3((CNKV * CHD) / 64, M / 128), blk, 0, stream>>>(xb, Wvt, vb, M, CNKV * CHD, CHID);

  // RoPE + bf16 convert
  int qtot = M * CNH * CHD, ktot = M * CNKV * CHD;
  rope_cvt<<<qtot / 256, blk, 0, stream>>>(qf, cosT, sinT, qb, CNH, qtot);
  rope_cvt<<<ktot / 256, blk, 0, stream>>>(kf, cosT, sinT, kb, CNKV, ktot);

  // Flash attention (writes bf16 into ab; reads only qb/kb/vb)
  attn_flash<<<dim3(CS / 128, CNH, CB), blk, 0, stream>>>(qb, kb, vb, ab);

  // Output projection
  gemm_bf16w<false><<<dim3(CHID / 64, M / 128), blk, 0, stream>>>(ab, Wot, out, M, CHID, CHID);
}
